// ModelForEntityCoreference_24756191494420
// MI455X (gfx1250) — matrix operand order repaired and checked by reading, not compiled, not run on hardware
//
#include <hip/hip_runtime.h>

// ---------------------------------------------------------------------------
// Types for WMMA (CDNA5 / gfx1250, wave32)
// ---------------------------------------------------------------------------
typedef __bf16 bf16_t;
typedef bf16_t v16bf __attribute__((ext_vector_type(16)));
typedef float  v8f   __attribute__((ext_vector_type(8)));
typedef unsigned int u32;
typedef u32    v4u   __attribute__((ext_vector_type(4)));

union FragAB { v4u u4[2]; v16bf v; };

__device__ __forceinline__ unsigned short f2bf(float f) {
  u32 u = __float_as_uint(f);
  u += 0x7fffu + ((u >> 16) & 1u);      // round-to-nearest-even
  return (unsigned short)(u >> 16);
}

__device__ __forceinline__ void wave_lds_fence() {
  asm volatile("s_wait_dscnt 0x0" ::: "memory");
}

// Async copy 16B/lane from global to LDS (ASYNCcnt-tracked, gfx1250).
__device__ __forceinline__ void async_copy_b128(unsigned lds_off,
                                                unsigned long long gaddr) {
  asm volatile("global_load_async_to_lds_b128 %0, %1, off"
               :: "v"(lds_off), "v"(gaddr) : "memory");
}

// Problem constants
#define BB 4
#define UU 16
#define LL 128
#define MM 256
#define HH 768
#define HID 150
#define NP 160                 // HID padded to 10 WMMA n-tiles
#define PPB 32640              // pairs per batch = M*(M-1)/2
#define TPB 2040               // 16-row tiles per batch
#define NTILES 10              // NP/16
#define KSTEPS1 24             // 768/32
#define KSTEPS2 5              // 160/32
#define NCHUNK 12              // K chunks of 64 (2 k-steps)
#define CHUNK_V4 1280          // 20480 B per chunk in v4u units

// ---------------------------------------------------------------------------
// 1) Span mean pooling: pooled[b,m,h]
// ---------------------------------------------------------------------------
__global__ __launch_bounds__(256) void pool_kernel(
    const float* __restrict__ hidden, const int* __restrict__ sutt,
    const int* __restrict__ sst, const int* __restrict__ sen,
    float* __restrict__ pooledF) {
  int bm = blockIdx.x;                 // b*M + m
  int b  = bm >> 8;
  int utt = sutt[bm], st = sst[bm], en = sen[bm];
  float inv = 1.0f / (float)(en - st);
  size_t base = ((size_t)(b * UU + utt)) * LL * HH;
  for (int h = threadIdx.x; h < HH; h += 256) {
    float a = 0.f;
    for (int pos = st; pos < en; ++pos)
      a += hidden[base + (size_t)pos * HH + h];
    pooledF[(size_t)bm * HH + h] = a * inv;
  }
}

// ---------------------------------------------------------------------------
// 2) hi = pooled @ W1a, hj = pooled @ W1b   (f32, padded to NP, pad = 0)
// ---------------------------------------------------------------------------
__global__ __launch_bounds__(256) void hihj_kernel(
    const float* __restrict__ pooledF, const float* __restrict__ W1,
    float* __restrict__ hiG, float* __restrict__ hjG) {
  __shared__ float sh[HH];
  int bm = blockIdx.x;
  for (int k = threadIdx.x; k < HH; k += 256)
    sh[k] = pooledF[(size_t)bm * HH + k];
  __syncthreads();
  for (int o = threadIdx.x; o < 2 * NP; o += 256) {
    int part = o / NP;                 // 0 -> hi, 1 -> hj
    int n = o - part * NP;
    float acc = 0.f;
    if (n < HID) {
      const float* w = W1 + (size_t)(part ? HH : 0) * HID + n;
      #pragma unroll 4
      for (int k = 0; k < HH; ++k) acc += sh[k] * w[(size_t)k * HID];
    }
    float* dst = part ? hjG : hiG;
    dst[(size_t)bm * NP + n] = acc;
  }
}

// ---------------------------------------------------------------------------
// 3) Pre-swizzle W1c (rows 2H..3H of W1) into per-lane B-fragment layout.
//    Layout: [kstep][ntile][lane][8 dwords]; dword v of lane l holds bf16
//    pair (k0,n),(k0+1,n), k0 = kstep*32 + (l>=16?16:0) + 2v, n = nt*16+l%16.
// ---------------------------------------------------------------------------
__global__ __launch_bounds__(256) void prep_w1csw(
    const float* __restrict__ W1, u32* __restrict__ out) {
  int idx = blockIdx.x * 256 + threadIdx.x;            // < 61440
  int v = idx & 7;
  int lane = (idx >> 3) & 31;
  int fi = idx >> 8;                                   // kstep*10+nt
  int nt = fi % NTILES, kstep = fi / NTILES;
  int n  = nt * 16 + (lane & 15);
  int k0 = kstep * 32 + ((lane >> 4) << 4) + 2 * v;
  unsigned short h0 = 0, h1 = 0;
  if (n < HID) {
    h0 = f2bf(W1[(size_t)(2 * HH + k0) * HID + n]);
    h1 = f2bf(W1[(size_t)(2 * HH + k0 + 1) * HID + n]);
  }
  out[idx] = (u32)h0 | ((u32)h1 << 16);
}

__global__ __launch_bounds__(256) void prep_w2sw(
    const float* __restrict__ W2, u32* __restrict__ out) {
  int idx = blockIdx.x * 256 + threadIdx.x;            // < 12800
  int v = idx & 7;
  int lane = (idx >> 3) & 31;
  int fi = idx >> 8;
  int nt = fi % NTILES, kstep = fi / NTILES;
  int n  = nt * 16 + (lane & 15);
  int k0 = kstep * 32 + ((lane >> 4) << 4) + 2 * v;
  unsigned short h0 = 0, h1 = 0;
  if (n < HID && k0 < HID)     h0 = f2bf(W2[(size_t)k0 * HID + n]);
  if (n < HID && k0 + 1 < HID) h1 = f2bf(W2[(size_t)(k0 + 1) * HID + n]);
  out[idx] = (u32)h0 | ((u32)h1 << 16);
}

__global__ void prep_pairs(int* __restrict__ iiA, int* __restrict__ jjA) {
  int i = blockIdx.x + 1;              // 1..255
  int j = threadIdx.x;
  if (j < i) {
    int p = (i * (i - 1)) / 2 + j;
    iiA[p] = i; jjA[p] = j;
  }
}

// ---------------------------------------------------------------------------
// 4) Main fused pair kernel.  1 wave = 16 pairs; 4 waves/block.
//    GEMM1 (prod @ W1c): W1c streamed into a double-buffered LDS chunk with
//    global_load_async_to_lds_b128 (overlapped with WMMA), fused +hi+hj+b1,
//    ReLU; GEMM2 (h1 @ W2) with W2 fragments from L2, fused +b2, ReLU,
//    dot W3, scatter score into logits[b][i][j].
// ---------------------------------------------------------------------------
__global__ __launch_bounds__(128) void coref_pair_kernel(
    const float* __restrict__ pooledF, const float* __restrict__ hiG,
    const float* __restrict__ hjG, const v4u* __restrict__ w1csw,
    const v4u* __restrict__ w2sw, const int* __restrict__ iiA,
    const int* __restrict__ jjA, const float* __restrict__ b1,
    const float* __restrict__ b2, const float* __restrict__ W3,
    const float* __restrict__ b3, float* __restrict__ logits) {
  // 2 x 20KB W1c chunk buffers + 4 x 5KB wave regions = 60KB
  __shared__ v4u smem[3840];

  const int tid  = threadIdx.x;
  const int wid  = tid >> 5;
  const int lane = tid & 31;
  const int mloc = lane & 15;
  const int hi16 = lane >> 4;          // 0 or 1

  const int rt = blockIdx.x * 4 + wid; // 0..8159 row tiles
  const int b  = rt / TPB;
  const int t  = rt - b * TPB;
  const int myP = t * 16 + mloc;
  const int iv = iiA[myP];
  const int jv = jjA[myP];

  unsigned short* wreg = (unsigned short*)((char*)smem + 40960 + wid * 5120);
  const v4u* wregU4 = (const v4u*)wreg;
  const unsigned lds_base = (unsigned)(uintptr_t)(void*)smem;
  const unsigned long long gsrc_base =
      (unsigned long long)(uintptr_t)(const void*)w1csw;

  v8f zero = {0.f, 0.f, 0.f, 0.f, 0.f, 0.f, 0.f, 0.f};
  v8f acc[NTILES];
  #pragma unroll
  for (int nt = 0; nt < NTILES; ++nt) acc[nt] = zero;

  const size_t rowB = (size_t)b * MM;

  // ---------------- GEMM1: prod @ W1c (async double-buffered) ----------
  // prefetch chunk 0 into buffer 0
  #pragma unroll
  for (int k = 0; k < 10; ++k) {
    unsigned off = (unsigned)(tid + k * 128) * 16u;
    async_copy_b128(lds_base + off, gsrc_base + off);
  }
  for (int kc = 0; kc < NCHUNK; ++kc) {
    if (kc + 1 < NCHUNK) {             // prefetch next chunk into other buffer
      unsigned buf = ((unsigned)(kc + 1) & 1u) * 20480u;
      unsigned long long src = gsrc_base + (unsigned long long)(kc + 1) * 20480u;
      #pragma unroll
      for (int k = 0; k < 10; ++k) {
        unsigned off = (unsigned)(tid + k * 128) * 16u;
        async_copy_b128(lds_base + buf + off, src + off);
      }
      asm volatile("s_wait_asynccnt 0xa" ::: "memory");  // current chunk landed
    } else {
      asm volatile("s_wait_asynccnt 0x0" ::: "memory");
    }
    __syncthreads();                   // all waves' portions visible
    const v4u* bbuf = smem + ((unsigned)kc & 1u) * CHUNK_V4;
    for (int ks = 0; ks < 2; ++ks) {
      const int kbase = (kc * 2 + ks) * 32;
      // stage A tile (16 x 32 bf16) : prod = pooled[i] * pooled[j]
      #pragma unroll
      for (int m = 0; m < 16; ++m) {
        int im = __shfl(iv, m, 32);
        int jm = __shfl(jv, m, 32);
        float a = pooledF[(rowB + im) * HH + kbase + lane];
        float c = pooledF[(rowB + jm) * HH + kbase + lane];
        wreg[m * 32 + lane] = f2bf(a * c);
      }
      wave_lds_fence();
      FragAB aF;
      aF.u4[0] = wregU4[mloc * 4 + 2 * hi16];
      aF.u4[1] = wregU4[mloc * 4 + 2 * hi16 + 1];
      #pragma unroll
      for (int nt = 0; nt < NTILES; ++nt) {
        FragAB bF;
        int x = ((ks * NTILES + nt) * 32 + lane) * 2;
        bF.u4[0] = bbuf[x];
        bF.u4[1] = bbuf[x + 1];
        acc[nt] = __builtin_amdgcn_wmma_f32_16x16x32_bf16(
            false, aF.v, false, bF.v, (short)0, acc[nt], false, false);
      }
      wave_lds_fence();                // B/A loads done before restaging
    }
    __syncthreads();                   // buffer free for next async issue
  }

  // ---------------- epilogue 1: + hi + hj + b1, ReLU, stage h1 (bf16) ------
  const float b3v = b3[0];
  #pragma unroll
  for (int nt = 0; nt < NTILES; ++nt) {
    int n = nt * 16 + mloc;
    float b1v = (n < HID) ? b1[n] : 0.f;
    #pragma unroll
    for (int r = 0; r < 8; ++r) {
      int m = r + 8 * hi16;
      int imr = __shfl(iv, m, 32);
      int jmr = __shfl(jv, m, 32);
      float v = acc[nt][r] + b1v;
      if (n < HID)
        v += hiG[(rowB + imr) * NP + n] + hjG[(rowB + jmr) * NP + n];
      v = v > 0.f ? v : 0.f;
      wreg[m * NP + n] = f2bf(v);
    }
  }
  wave_lds_fence();

  // ---------------- GEMM2: h1 @ W2 ----------------
  v8f acc2[NTILES];
  #pragma unroll
  for (int nt = 0; nt < NTILES; ++nt) acc2[nt] = zero;
  for (int ks = 0; ks < KSTEPS2; ++ks) {
    FragAB aF;
    aF.u4[0] = wregU4[mloc * 20 + ks * 4 + 2 * hi16];
    aF.u4[1] = wregU4[mloc * 20 + ks * 4 + 2 * hi16 + 1];
    #pragma unroll
    for (int nt = 0; nt < NTILES; ++nt) {
      FragAB bF;
      size_t x = (size_t)((ks * NTILES + nt) * 32 + lane) * 2;
      bF.u4[0] = w2sw[x];
      bF.u4[1] = w2sw[x + 1];
      acc2[nt] = __builtin_amdgcn_wmma_f32_16x16x32_bf16(
          false, aF.v, false, bF.v, (short)0, acc2[nt], false, false);
    }
  }

  // ---------------- epilogue 2: +b2, ReLU, dot W3, scatter score ----------
  float w3l[NTILES], b2l[NTILES];
  #pragma unroll
  for (int nt = 0; nt < NTILES; ++nt) {
    int n = nt * 16 + mloc;
    w3l[nt] = (n < HID) ? W3[n] : 0.f;
    b2l[nt] = (n < HID) ? b2[n] : 0.f;
  }
  float s[8];
  #pragma unroll
  for (int r = 0; r < 8; ++r) {
    float p = 0.f;
    #pragma unroll
    for (int nt = 0; nt < NTILES; ++nt) {
      float h = acc2[nt][r] + b2l[nt];
      h = h > 0.f ? h : 0.f;
      p += h * w3l[nt];
    }
    p += __shfl_xor(p, 1, 32);
    p += __shfl_xor(p, 2, 32);
    p += __shfl_xor(p, 4, 32);
    p += __shfl_xor(p, 8, 32);
    s[r] = p + b3v;
  }
  #pragma unroll
  for (int r = 0; r < 8; ++r) {
    int m = r + 8 * hi16;
    int imr = __shfl(iv, m, 32);
    int jmr = __shfl(jv, m, 32);
    if (mloc == 0)
      logits[(rowB + imr) * MM + jmr] = s[r];
  }
}

// ---------------------------------------------------------------------------
// 5) Per-row masked softmax + clipped label-sum + -log
// ---------------------------------------------------------------------------
__global__ __launch_bounds__(256) void loss_rows(
    const float* __restrict__ logits, const float* __restrict__ labels,
    float* __restrict__ rowloss) {
  __shared__ float red[256];
  int bi = blockIdx.x;                 // b*M + i
  int i  = bi & 255;
  int j  = threadIdx.x;
  const float* lr = logits + (size_t)bi * MM;
  bool valid = (j <= i);
  float lx = valid ? lr[j] : -1e30f;
  red[j] = lx; __syncthreads();
  for (int sft = 128; sft > 0; sft >>= 1) {
    if (j < sft) red[j] = fmaxf(red[j], red[j + sft]);
    __syncthreads();
  }
  float mx = red[0]; __syncthreads();
  float ex = valid ? expf(lx - mx) : 0.f;
  red[j] = ex; __syncthreads();
  for (int sft = 128; sft > 0; sft >>= 1) {
    if (j < sft) red[j] += red[j + sft];
    __syncthreads();
  }
  float ssum = red[0]; __syncthreads();
  float p = valid ? (ex / ssum) : -1000.f;
  float lab = labels[(size_t)bi * MM + j];
  float term = fminf(fmaxf(p * lab, 1e-8f), 1.f - 1e-8f);
  red[j] = term; __syncthreads();
  for (int sft = 128; sft > 0; sft >>= 1) {
    if (j < sft) red[j] += red[j + sft];
    __syncthreads();
  }
  if (j == 0) rowloss[bi] = -logf(red[0]);
}

__global__ __launch_bounds__(256) void loss_final(
    const float* __restrict__ rowloss, float* __restrict__ out) {
  __shared__ float red[256];
  int t = threadIdx.x;
  red[t] = rowloss[t] + rowloss[t + 256] + rowloss[t + 512] + rowloss[t + 768];
  __syncthreads();
  for (int sft = 128; sft > 0; sft >>= 1) {
    if (t < sft) red[t] += red[t + sft];
    __syncthreads();
  }
  if (t == 0) out[0] = red[0];
}

// ---------------------------------------------------------------------------
// Host launcher
// ---------------------------------------------------------------------------
extern "C" void kernel_launch(void* const* d_in, const int* in_sizes, int n_in,
                              void* d_out, int out_size, void* d_ws, size_t ws_size,
                              hipStream_t stream) {
  const float* hidden = (const float*)d_in[0];
  const int*   sutt   = (const int*)d_in[1];
  const int*   sst    = (const int*)d_in[2];
  const int*   sen    = (const int*)d_in[3];
  const float* labels = (const float*)d_in[4];
  const float* W1     = (const float*)d_in[5];
  const float* b1     = (const float*)d_in[6];
  const float* W2     = (const float*)d_in[7];
  const float* b2     = (const float*)d_in[8];
  const float* W3     = (const float*)d_in[9];
  const float* b3     = (const float*)d_in[10];
  float* out = (float*)d_out;

  char* ws = (char*)d_ws;
  constexpr size_t o_pool = 0;                         // 3,145,728 B
  constexpr size_t o_hi   = o_pool + (size_t)BB * MM * HH * 4;
  constexpr size_t o_hj   = o_hi   + (size_t)BB * MM * NP * 4;
  constexpr size_t o_w1   = o_hj   + (size_t)BB * MM * NP * 4;
  constexpr size_t o_w2   = o_w1   + (size_t)KSTEPS1 * NTILES * 32 * 8 * 4;
  constexpr size_t o_ii   = o_w2   + (size_t)KSTEPS2 * NTILES * 32 * 8 * 4;
  constexpr size_t o_jj   = o_ii   + (size_t)PPB * 4;
  constexpr size_t o_lg   = o_jj   + (size_t)PPB * 4;
  constexpr size_t o_rl   = o_lg   + (size_t)BB * MM * MM * 4;

  float* pooledF = (float*)(ws + o_pool);
  float* hiG     = (float*)(ws + o_hi);
  float* hjG     = (float*)(ws + o_hj);
  u32*   w1csw   = (u32*)(ws + o_w1);
  u32*   w2sw    = (u32*)(ws + o_w2);
  int*   iiA     = (int*)(ws + o_ii);
  int*   jjA     = (int*)(ws + o_jj);
  float* logits  = (float*)(ws + o_lg);
  float* rowloss = (float*)(ws + o_rl);

  pool_kernel<<<BB * MM, 256, 0, stream>>>(hidden, sutt, sst, sen, pooledF);
  hihj_kernel<<<BB * MM, 256, 0, stream>>>(pooledF, W1, hiG, hjG);
  prep_w1csw<<<240, 256, 0, stream>>>(W1, w1csw);
  prep_w2sw<<<50, 256, 0, stream>>>(W2, w2sw);
  prep_pairs<<<MM - 1, 256, 0, stream>>>(iiA, jjA);
  hipMemsetAsync(logits, 0, (size_t)BB * MM * MM * 4, stream);
  coref_pair_kernel<<<TPB, 128, 0, stream>>>(
      pooledF, hiG, hjG, (const v4u*)w1csw, (const v4u*)w2sw, iiA, jjA,
      b1, b2, W3, b3, logits);
  loss_rows<<<BB * MM, 256, 0, stream>>>(logits, labels, rowloss);
  loss_final<<<1, 256, 0, stream>>>(rowloss, out);
}
